// SplineCouplingLayer_13821204759266
// MI455X (gfx1250) — matrix operand order repaired and checked by reading, not compiled, not run on hardware
//
#include <hip/hip_runtime.h>

typedef __attribute__((ext_vector_type(16))) _Float16 v16h;
typedef __attribute__((ext_vector_type(8)))  float    v8f;

#define DIMX     64
#define D_INC    32
#define D_OUTC   32
#define HIDDEN   512
#define N_BINS   8
#define OUT_DIM  (D_OUTC * (3 * N_BINS - 1))   // 736
#define TAILB    3.0f

#define B_ROWS   32            // rows per workgroup
#define HS       520           // LDS stride (halfs) for h1/h2
#define RSTR     740           // LDS stride (floats) for raw
#define NT3      (OUT_DIM/16)  // 46
#define KT_H     (HIDDEN/32)   // 16 k-steps of 32

// ---- LDS partition (dynamic LDS, ~163.5 KB total) ----
#define RAW_BYTES   (B_ROWS * RSTR * 4)            // 94720
#define H1_OFF      (RAW_BYTES)                    // f16, B_ROWS*HS
#define H2_OFF      (H1_OFF + B_ROWS * HS * 2)     // f16, B_ROWS*HS
#define XM_OFF      (H2_OFF + B_ROWS * HS * 2)     // f16, B_ROWS*D_INC
#define LD_OFF      (XM_OFF + B_ROWS * D_INC * 2)  // f32, B_ROWS
#define SMEM_BYTES  (LD_OFF + B_ROWS * 4)          // 163456

// ---------------- prologue: f32 weights -> f16 in workspace ----------------
__global__ void cvt_weights_f16(const float* __restrict__ W1,
                                const float* __restrict__ W2,
                                const float* __restrict__ W3,
                                _Float16* __restrict__ W1h,
                                _Float16* __restrict__ W2h,
                                _Float16* __restrict__ W3h) {
  const int stride = gridDim.x * blockDim.x;
  const int i = blockIdx.x * blockDim.x + threadIdx.x;
  const int n1 = D_INC * HIDDEN, n2 = HIDDEN * HIDDEN, n3 = HIDDEN * OUT_DIM;
  for (int p = i; p < n1; p += stride) W1h[p] = (_Float16)W1[p];
  for (int p = i; p < n2; p += stride) W2h[p] = (_Float16)W2[p];
  for (int p = i; p < n3; p += stride) W3h[p] = (_Float16)W3[p];
}

// ---------------- WMMA fragment helpers (CDNA5 wave32 layouts) ----------------
__device__ __forceinline__ v8f wmma16(v16h a, v16h b, v8f c) {
  // v_wmma_f32_16x16x32_f16: 8 args (neg_a, A, neg_b, B, c_mod, C, reuse_a, reuse_b)
  return __builtin_amdgcn_wmma_f32_16x16x32_f16(false, a, false, b, (short)0, c,
                                                false, false);
}

// A-matrix 16x32 f16 from LDS (row-major, stride in halfs).
// Lane l: M = l&15, Kbase = (l>>4)*8; VGPR v<4 -> K=Kbase+2v, v>=4 -> K=Kbase+16+2(v-4).
__device__ __forceinline__ v16h load_a_lds(const _Float16* base, int stride, int kcol) {
  const int lane = threadIdx.x & 31;
  const int r  = lane & 15;
  const int kb = (lane >> 4) << 3;
  const _Float16* p = base + r * stride + kcol;
  union { v16h v; unsigned u[8]; } f;
#pragma unroll
  for (int i = 0; i < 8; ++i) {
    const int k = (i < 4) ? (kb + 2 * i) : (kb + 8 + 2 * i); // i>=4: kb+16+2(i-4)
    f.u[i] = *(const unsigned*)(p + k);
  }
  return f.v;
}

// B-matrix 32x16 f16 from global row-major W[K][ncols].
// Lane l: N = l&15, Kbase = (l>>4)*8; elem s<8 -> K=Kbase+s, s>=8 -> K=Kbase+16+(s-8).
__device__ __forceinline__ v16h load_b_glob(const _Float16* __restrict__ W, int ncols,
                                            int krow, int ncol) {
  const int lane = threadIdx.x & 31;
  const int n  = ncol + (lane & 15);
  const int kb = (lane >> 4) << 3;
  v16h b;
#pragma unroll
  for (int s = 0; s < 16; ++s) {
    const int k = krow + ((s < 8) ? (kb + s) : (kb + 8 + s));
    b[s] = W[(size_t)k * ncols + n];
  }
  return b;
}

// C/D f32 16x16: lane l -> N = l&15; VGPR r -> M = r + 8*(l>>4).
__device__ __forceinline__ void store_relu_f16(_Float16* dst, int stride, int mt, int nt,
                                               v8f c, const float* __restrict__ bias) {
  const int lane = threadIdx.x & 31;
  const int col  = lane & 15;
  const int rh   = (lane >> 4) << 3;
  const int n    = nt * 16 + col;
  const float bn = bias[n];
#pragma unroll
  for (int r = 0; r < 8; ++r) {
    float v = c[r] + bn;
    v = v > 0.0f ? v : 0.0f;
    dst[(mt * 16 + rh + r) * stride + n] = (_Float16)v;
  }
}

__device__ __forceinline__ void store_bias_f32(float* dst, int stride, int mt, int nt,
                                               v8f c, const float* __restrict__ bias) {
  const int lane = threadIdx.x & 31;
  const int col  = lane & 15;
  const int rh   = (lane >> 4) << 3;
  const int n    = nt * 16 + col;
  const float bn = bias[n];
#pragma unroll
  for (int r = 0; r < 8; ++r)
    dst[(mt * 16 + rh + r) * stride + n] = c[r] + bn;
}

// ---------------- fused MLP + RQ-spline kernel ----------------
__global__ __launch_bounds__(256, 1)
void spline_coupling_fused(const float* __restrict__ x,
                           const _Float16* __restrict__ W1h, const float* __restrict__ b1,
                           const _Float16* __restrict__ W2h, const float* __restrict__ b2,
                           const _Float16* __restrict__ W3h, const float* __restrict__ b3,
                           float* __restrict__ out, int nrows_total) {
  extern __shared__ __align__(16) char smem[];
  float*    sRaw = (float*)smem;                 // [B_ROWS][RSTR]
  _Float16* sH1  = (_Float16*)(smem + H1_OFF);   // [B_ROWS][HS]
  _Float16* sH2  = (_Float16*)(smem + H2_OFF);   // [B_ROWS][HS]
  _Float16* sXm  = (_Float16*)(smem + XM_OFF);   // [B_ROWS][D_INC]
  float*    sLd  = (float*)(smem + LD_OFF);      // [B_ROWS]

  const int tid  = threadIdx.x;
  const int lane = tid & 31;
  const int wv   = tid >> 5;
  const int row0 = blockIdx.x * B_ROWS;

  // ---- stage 0: x_masked -> f16 LDS, zero log-det accumulators
  for (int p = tid; p < B_ROWS * D_INC; p += 256) {
    const int r = p >> 5, c = p & 31;
    sXm[r * D_INC + c] = (_Float16)x[(size_t)(row0 + r) * DIMX + c];
  }
  if (tid < B_ROWS) sLd[tid] = 0.0f;
  __syncthreads();

  // ---- GEMM1: h1 = relu(xm @ W1 + b1); K = 32 = one WMMA per tile
  {
    const int mt  = wv >> 2;
    const int nt0 = (wv & 3) * 8;
    const v16h a = load_a_lds(sXm + mt * 16 * D_INC, D_INC, 0);
#pragma unroll
    for (int j = 0; j < 8; ++j) {
      const int nt = nt0 + j;
      v8f c = {};
      c = wmma16(a, load_b_glob(W1h, HIDDEN, 0, nt * 16), c);
      store_relu_f16(sH1, HS, mt, nt, c, b1);
    }
  }
  __syncthreads();

  // ---- GEMM2: h2 = relu(h1 @ W2 + b2)
  {
    const int mt  = wv >> 2;
    const int nt0 = (wv & 3) * 8;
    v8f acc[8];
    v8f z = {};
#pragma unroll
    for (int j = 0; j < 8; ++j) acc[j] = z;
    for (int kt = 0; kt < KT_H; ++kt) {
      if (kt + 1 < KT_H) { // pull next k-slab of W2 toward L0/L2 (global_prefetch_b8)
        const int kn = (kt + 1) * 32 + (lane & 15) + ((lane >> 4) << 4);
        __builtin_prefetch(W2h + (size_t)kn * HIDDEN + nt0 * 16, 0, 1);
      }
      const v16h a = load_a_lds(sH1 + mt * 16 * HS, HS, kt * 32);
#pragma unroll
      for (int j = 0; j < 8; ++j)
        acc[j] = wmma16(a, load_b_glob(W2h, HIDDEN, kt * 32, (nt0 + j) * 16), acc[j]);
    }
#pragma unroll
    for (int j = 0; j < 8; ++j) store_relu_f16(sH2, HS, mt, nt0 + j, acc[j], b2);
  }
  __syncthreads();

  // ---- GEMM3: raw = h2 @ W3 + b3   (46 N-tiles; B-fragment reused over both M-tiles)
  for (int nt = wv; nt < NT3; nt += 8) {
    v8f a0 = {}, a1 = {};
    for (int kt = 0; kt < KT_H; ++kt) {
      const v16h bf  = load_b_glob(W3h, OUT_DIM, kt * 32, nt * 16);
      const v16h af0 = load_a_lds(sH2,            HS, kt * 32);
      const v16h af1 = load_a_lds(sH2 + 16 * HS,  HS, kt * 32);
      a0 = wmma16(af0, bf, a0);
      a1 = wmma16(af1, bf, a1);
    }
    store_bias_f32(sRaw, RSTR, 0, nt, a0, b3);
    store_bias_f32(sRaw, RSTR, 1, nt, a1, b3);
  }
  __syncthreads();

  // ---- RQ spline per (row, channel)
  for (int p = tid; p < B_ROWS * D_OUTC; p += 256) {
    const int r = p >> 5, ch = p & 31;
    const size_t grow = (size_t)(row0 + r);

    // passthrough half
    out[grow * DIMX + ch] = x[grow * DIMX + ch];
    const float xt = x[grow * DIMX + D_INC + ch];

    const float* rp = sRaw + r * RSTR + ch * (3 * N_BINS - 1);

    // softmax(widths)*6, softmax(heights)*6
    float wr[N_BINS], hr[N_BINS];
#pragma unroll
    for (int k = 0; k < N_BINS; ++k) { wr[k] = rp[k]; hr[k] = rp[N_BINS + k]; }
    float mw = wr[0], mh = hr[0];
#pragma unroll
    for (int k = 1; k < N_BINS; ++k) { mw = fmaxf(mw, wr[k]); mh = fmaxf(mh, hr[k]); }
    float sw = 0.0f, sh = 0.0f;
#pragma unroll
    for (int k = 0; k < N_BINS; ++k) {
      wr[k] = expf(wr[k] - mw); sw += wr[k];
      hr[k] = expf(hr[k] - mh); sh += hr[k];
    }
    const float cww = (2.0f * TAILB) / sw, chs = (2.0f * TAILB) / sh;
#pragma unroll
    for (int k = 0; k < N_BINS; ++k) { wr[k] *= cww; hr[k] *= chs; }

    // derivatives = [1, softplus(raw), 1]
    float der[N_BINS + 1];
    der[0] = 1.0f; der[N_BINS] = 1.0f;
#pragma unroll
    for (int k = 0; k < N_BINS - 1; ++k) {
      const float d = rp[2 * N_BINS + k];
      der[k + 1] = (d > 20.0f) ? d : log1pf(expf(d));
    }

    // cumulative widths/heights starting at -TAIL
    float cw[N_BINS + 1], chv[N_BINS + 1];
    cw[0] = -TAILB; chv[0] = -TAILB;
#pragma unroll
    for (int k = 0; k < N_BINS; ++k) {
      cw[k + 1]  = cw[k]  + wr[k];
      chv[k + 1] = chv[k] + hr[k];
    }

    const bool inside = (xt >= -TAILB) && (xt <= TAILB);
    const float xc = fminf(fmaxf(xt, -TAILB + 1e-6f), TAILB - 1e-6f);

    int idx = 0;
#pragma unroll
    for (int k = 0; k < N_BINS; ++k) idx += (cw[k + 1] < xc) ? 1 : 0;
    idx = (idx > N_BINS - 1) ? (N_BINS - 1) : idx;

    // gather via unrolled selects (stays in VGPRs, no scratch)
    float w_k = 0.f, h_k = 0.f, cw_k = 0.f, ch_k = 0.f, d_k = 0.f, d_k1 = 0.f;
#pragma unroll
    for (int k = 0; k < N_BINS; ++k) {
      const bool m = (k == idx);
      w_k  = m ? wr[k]      : w_k;
      h_k  = m ? hr[k]      : h_k;
      cw_k = m ? cw[k]      : cw_k;
      ch_k = m ? chv[k]     : ch_k;
      d_k  = m ? der[k]     : d_k;
      d_k1 = m ? der[k + 1] : d_k1;
    }

    float xi = (xc - cw_k) / w_k;
    xi = fminf(fmaxf(xi, 1e-6f), 1.0f - 1e-6f);
    const float om  = 1.0f - xi;
    const float sk  = h_k / w_k;
    const float num = h_k * (sk * xi * xi + d_k * xi * om);
    const float den = sk + (d_k + d_k1 - 2.0f * sk) * xi * om;
    const float y_in  = ch_k + num / den;
    const float ld_in =
        logf(sk * sk * (d_k1 * xi * xi + 2.0f * sk * xi * om + d_k * om * om) + 1e-8f) -
        logf(den * den + 1e-8f);

    out[grow * DIMX + D_INC + ch] = inside ? y_in : xt;
    atomicAdd(&sLd[r], inside ? ld_in : 0.0f);   // ds_add_f32
  }
  __syncthreads();

  // ---- log-det output
  for (int r = tid; r < B_ROWS; r += 256)
    out[(size_t)nrows_total * DIMX + row0 + r] = sLd[r];
}

// ---------------- launch ----------------
extern "C" void kernel_launch(void* const* d_in, const int* in_sizes, int n_in,
                              void* d_out, int out_size, void* d_ws, size_t ws_size,
                              hipStream_t stream) {
  const float* x  = (const float*)d_in[0];
  const float* W1 = (const float*)d_in[1];
  const float* b1 = (const float*)d_in[2];
  const float* W2 = (const float*)d_in[3];
  const float* b2 = (const float*)d_in[4];
  const float* W3 = (const float*)d_in[5];
  const float* b3 = (const float*)d_in[6];
  float* out = (float*)d_out;

  const int nrows = in_sizes[0] / DIMX;

  _Float16* W1h = (_Float16*)d_ws;
  _Float16* W2h = W1h + D_INC * HIDDEN;
  _Float16* W3h = W2h + HIDDEN * HIDDEN;

  cvt_weights_f16<<<256, 256, 0, stream>>>(W1, W2, W3, W1h, W2h, W3h);

  spline_coupling_fused<<<nrows / B_ROWS, 256, SMEM_BYTES, stream>>>(
      x, W1h, b1, W2h, b2, W3h, b3, out, nrows);
}
